// SimpleRNN_13675175871073
// MI455X (gfx1250) — matrix operand order repaired and checked by reading, not compiled, not run on hardware
//
#include <hip/hip_runtime.h>
#include <hip/hip_bf16.h>
#include <math.h>

// Problem constants (reference: B=64, S=2048, I=H=O=256)
#define BB 64
#define SS 2048
#define HH 256
#define LDSW 264   // padded LDS row stride in bf16 elements (256 + 8)

typedef __attribute__((ext_vector_type(16))) __bf16 v16bf;
typedef __attribute__((ext_vector_type(8)))  __bf16 v8bf;
typedef __attribute__((ext_vector_type(8)))  float  v8f;

union BF16V { v16bf v; v8bf h[2]; };

static __device__ __forceinline__ __bf16 f2bf(float f) {
  unsigned u = __builtin_bit_cast(unsigned, f);
  unsigned r = (u + 0x7FFFu + ((u >> 16) & 1u)) >> 16;     // round-to-nearest-even
  return __builtin_bit_cast(__bf16, (unsigned short)r);
}
static __device__ __forceinline__ float bf2f(__bf16 b) {
  unsigned short s = __builtin_bit_cast(unsigned short, b);
  return __builtin_bit_cast(float, (unsigned)s << 16);
}

// Hardware tanh (CDNA5 V_TANH_F32 trans op) instead of branchy libm tanhf.
static __device__ __forceinline__ float fast_tanh(float x) {
#if __has_builtin(__builtin_amdgcn_tanhf)
  return __builtin_amdgcn_tanhf(x);
#else
  float r;
  asm volatile("v_tanh_f32 %0, %1" : "=v"(r) : "v"(x));
  return r;
#endif
}

// ---------------------------------------------------------------------------
// Generic  C[M,256] = A[M,256] @ W[256,256]^T + bias   (W row-major [n][k])
// W staged once per block into LDS as bf16 (padded rows). 8 waves; each wave
// computes a 16(M) x 256(N) strip with v_wmma_f32_16x16x32_bf16.
// Row remap for output: r = p<<din_shift | q  ->  r_out = q<<dout_shift | p
// ---------------------------------------------------------------------------
template <bool A_F32, bool OUT_F32>
__global__ void __launch_bounds__(256)
gemm256(const void* __restrict__ Aptr, const float* __restrict__ W,
        const float* __restrict__ bias, void* __restrict__ Out,
        int din_shift, int dout_shift)
{
  extern __shared__ char smem[];
  __bf16* Wl    = (__bf16*)smem;                        // 256 x LDSW bf16
  float*  biasl = (float*)(smem + 256 * LDSW * 2);      // 256 f32

  const int tid = threadIdx.x;
  for (int i = tid; i < 256 * 256; i += 256)
    Wl[(i >> 8) * LDSW + (i & 255)] = f2bf(W[i]);
  biasl[tid] = bias[tid];
  __syncthreads();

  const int wave = tid >> 5, lane = tid & 31;
  const int half = lane >> 4, lanelo = lane & 15;
  const int mbase = blockIdx.x * 128 + wave * 16;
  const size_t arow = (size_t)(mbase + lanelo) * 256;

  v8f c[16];
#pragma unroll
  for (int nt = 0; nt < 16; ++nt) {
    float bv = biasl[nt * 16 + lanelo];
#pragma unroll
    for (int v = 0; v < 8; ++v) c[nt][v] = bv;
  }

#pragma unroll
  for (int kb = 0; kb < 8; ++kb) {
    const int k0 = kb * 32 + 8 * half;   // A layout: K = k0+{0..7} and k0+16+{0..7}
    BF16V a;
    if constexpr (A_F32) {
      const float* Af = (const float*)Aptr + arow;
      v8f g0 = *(const v8f*)(Af + k0);
      v8f g1 = *(const v8f*)(Af + k0 + 16);
#pragma unroll
      for (int i = 0; i < 8; ++i) { a.v[i] = f2bf(g0[i]); a.v[i + 8] = f2bf(g1[i]); }
    } else {
      const __bf16* Ab = (const __bf16*)Aptr + arow;
      a.h[0] = *(const v8bf*)(Ab + k0);
      a.h[1] = *(const v8bf*)(Ab + k0 + 16);
    }
#pragma unroll
    for (int nt = 0; nt < 16; ++nt) {
      const __bf16* bp = Wl + (nt * 16 + lanelo) * LDSW + kb * 32 + 8 * half;
      BF16V b;
      b.h[0] = *(const v8bf*)bp;
      b.h[1] = *(const v8bf*)(bp + 16);
      c[nt] = __builtin_amdgcn_wmma_f32_16x16x32_bf16(false, a.v, false, b.v,
                                                      (short)0, c[nt], false, false);
    }
  }

  const int dmask = (1 << din_shift) - 1;
#pragma unroll
  for (int nt = 0; nt < 16; ++nt) {
    const int n = nt * 16 + lanelo;
#pragma unroll
    for (int v = 0; v < 8; ++v) {
      int r = mbase + v + 8 * half;
      size_t ro = ((size_t)(r & dmask) << dout_shift) + (r >> din_shift);
      if constexpr (OUT_F32) ((float*)Out)[ro * 256 + n] = c[nt][v];
      else                   ((__bf16*)Out)[ro * 256 + n] = f2bf(c[nt][v]);
    }
  }
}

// ---------------------------------------------------------------------------
// Recurrent scan. 4 blocks x 256 threads; block owns 16 batch rows.
// Each wave owns 32 H-columns and holds its W_hh slice (16 KB) entirely in
// 128 VGPRs as preformatted WMMA-B operands -> the per-step chain touches LDS
// only for the A operand (h, double-buffered, ~17 KB LDS).
// Per step: h_new = tanh(xw[t] + h @ W_hh^T); h_new also streamed to h_all.
// ---------------------------------------------------------------------------
__global__ void __launch_bounds__(256)
rnn_scan(const __bf16* __restrict__ xw,    // [S,B,H] bf16 (includes b_h)
         const float*  __restrict__ Whh,   // [H,H] f32
         __bf16* __restrict__ h_all,       // [S,B,H] bf16
         float*  __restrict__ h_final)     // [B,H] f32
{
  extern __shared__ char smem[];
  __bf16* hb = (__bf16*)smem;              // 2 x 16 x LDSW bf16

  const int tid = threadIdx.x;
  for (int i = tid; i < 2 * 16 * LDSW; i += 256)
    hb[i] = __builtin_bit_cast(__bf16, (unsigned short)0);

  const int wave = tid >> 5, lane = tid & 31;
  const int half = lane >> 4, lanelo = lane & 15;
  const int bbase = blockIdx.x * 16;       // batch tile
  const int nb = wave * 32;                // this wave's H-column base

  // Preload this wave's W_hh slice as WMMA-B operands (loop-invariant).
  // B layout: lane -> N = nb + nt*16 + lanelo; element e -> K = kb*32 + 16*half + e.
  BF16V bw[8][2];
#pragma unroll
  for (int kb = 0; kb < 8; ++kb) {
#pragma unroll
    for (int nt = 0; nt < 2; ++nt) {
      const int n = nb + nt * 16 + lanelo;
      const float* wp = Whh + (size_t)n * HH + kb * 32 + 16 * half;
      v8f g0 = *(const v8f*)wp;
      v8f g1 = *(const v8f*)(wp + 8);
#pragma unroll
      for (int i = 0; i < 8; ++i) {
        bw[kb][nt].v[i]     = f2bf(g0[i]);
        bw[kb][nt].v[i + 8] = f2bf(g1[i]);
      }
    }
  }

  // software-pipelined xw prefetch (one step ahead)
  float xwv[2][8];
#pragma unroll
  for (int nt = 0; nt < 2; ++nt) {
    const int n = nb + nt * 16 + lanelo;
#pragma unroll
    for (int v = 0; v < 8; ++v)
      xwv[nt][v] = bf2f(xw[(size_t)(bbase + v + 8 * half) * HH + n]);
  }

  __syncthreads();   // h zero-init visible

  for (int t = 0; t < SS; ++t) {
    const __bf16* hcur = hb + (t & 1) * 16 * LDSW;
    __bf16*       hnxt = hb + ((t + 1) & 1) * 16 * LDSW;

    v8f c[2];
#pragma unroll
    for (int nt = 0; nt < 2; ++nt)
#pragma unroll
      for (int v = 0; v < 8; ++v) c[nt][v] = xwv[nt][v];

    // prefetch xw for t+1 (overlaps the WMMA chain below)
    if (t + 1 < SS) {
      const size_t tb = (size_t)(t + 1) * (BB * HH);
#pragma unroll
      for (int nt = 0; nt < 2; ++nt) {
        const int n = nb + nt * 16 + lanelo;
#pragma unroll
        for (int v = 0; v < 8; ++v)
          xwv[nt][v] = bf2f(xw[tb + (size_t)(bbase + v + 8 * half) * HH + n]);
      }
    }

#pragma unroll
    for (int kb = 0; kb < 8; ++kb) {
      BF16V a;
      const __bf16* ap = hcur + lanelo * LDSW + kb * 32 + 16 * half;
      a.h[0] = *(const v8bf*)ap;
      a.h[1] = *(const v8bf*)(ap + 8);
#pragma unroll
      for (int nt = 0; nt < 2; ++nt) {
        c[nt] = __builtin_amdgcn_wmma_f32_16x16x32_bf16(false, a.v, false, bw[kb][nt].v,
                                                        (short)0, c[nt], false, false);
      }
    }

    const size_t tbase = (size_t)t * (BB * HH);
#pragma unroll
    for (int nt = 0; nt < 2; ++nt) {
      const int n = nb + nt * 16 + lanelo;
#pragma unroll
      for (int v = 0; v < 8; ++v) {
        const int m = v + 8 * half;
        float hv = fast_tanh(c[nt][v]);
        __bf16 hbf = f2bf(hv);
        hnxt[m * LDSW + n] = hbf;
        h_all[tbase + (size_t)(bbase + m) * HH + n] = hbf;
        if (t == SS - 1) h_final[(size_t)(bbase + m) * HH + n] = hv;
      }
    }
    __syncthreads();
  }
}

// ---------------------------------------------------------------------------
extern "C" void kernel_launch(void* const* d_in, const int* in_sizes, int n_in,
                              void* d_out, int out_size, void* d_ws, size_t ws_size,
                              hipStream_t stream) {
  (void)in_sizes; (void)n_in; (void)out_size; (void)ws_size;
  const float* x    = (const float*)d_in[0];
  const float* W_ih = (const float*)d_in[1];
  const float* W_hh = (const float*)d_in[2];
  const float* b_h  = (const float*)d_in[3];
  const float* W_ho = (const float*)d_in[4];
  const float* b_o  = (const float*)d_in[5];

  __bf16* xw    = (__bf16*)d_ws;                       // [S,B,H] bf16: 64 MB
  __bf16* h_all = xw + (size_t)SS * BB * HH;           // [S,B,H] bf16: 64 MB
  float*  outputs = (float*)d_out;                     // [B,S,O]
  float*  h_final = outputs + (size_t)BB * SS * 256;   // [B,H]

  const int M = BB * SS;                               // 131072 rows
  const size_t gemm_lds = (size_t)256 * LDSW * 2 + 256 * 4;
  const size_t rnn_lds  = (size_t)2 * 16 * LDSW * 2;

  // Phase 1: xw[S,B,H] = x[B,S,I] @ W_ih^T + b_h   (rows r=b*2048+s -> s*64+b)
  gemm256<true, false><<<M / 128, 256, gemm_lds, stream>>>(x, W_ih, b_h, xw, 11, 6);

  // Phase 2: sequential scan (W_hh slice in VGPRs, h double-buffered in LDS)
  rnn_scan<<<BB / 16, 256, rnn_lds, stream>>>(xw, W_hh, h_all, h_final);

  // Phase 3: outputs[B,S,O] = h_all[S,B,H] @ W_ho^T + b_o (rows r=s*64+b -> b*2048+s)
  gemm256<false, true><<<M / 128, 256, gemm_lds, stream>>>(h_all, W_ho, b_o, outputs, 6, 11);
}
